// LatentMoELayer_12773232738935
// MI455X (gfx1250) — matrix operand order repaired and checked by reading, compiled, not run on hardware
//
#include <hip/hip_runtime.h>

#define DEV __device__ __forceinline__

typedef __attribute__((ext_vector_type(16))) __bf16 v16bf;
typedef __attribute__((ext_vector_type(8)))  float  v8f;
typedef __attribute__((ext_vector_type(4)))  unsigned int u32x4;
typedef __attribute__((ext_vector_type(8)))  int          i32x8;
typedef __attribute__((ext_vector_type(4)))  int          i32x4;

union FragB16 { uint4 q[2]; v16bf v; };
union Pack8   { uint4 q; unsigned short h[8]; };

DEV unsigned short f32_to_bf16(float f) {
  unsigned int u = __float_as_uint(f);
  u += 0x7fffu + ((u >> 16) & 1u);          // round-to-nearest-even
  return (unsigned short)(u >> 16);
}
DEV float silu_f(float g) { return g / (1.f + __expf(-g)); }

// ---- CDNA5 async global->LDS copy (ASYNCcnt path, CDNA5 ISA) ----
DEV unsigned int lds_off_u32(const void* p) {        // low 32 bits of generic = LDS offset
  return (unsigned int)(unsigned long long)p;
}
DEV void async_copy_g2l_b128(unsigned int ldsoff, const void* g) {
  asm volatile("global_load_async_to_lds_b128 %0, %1, off"
               :: "v"(ldsoff), "v"(g) : "memory");
}
#define WAIT_ASYNCCNT(imm) asm volatile("s_wait_asynccnt " imm ::: "memory")

// ---- CDNA5 Tensor Data Mover: 2D tile DMA global->LDS (TENSORcnt path, §7/§8) ----
// Loads a [rows x 32] bf16 tile (32 contiguous K elems per row, row stride = ld elems)
// row-major contiguous into LDS at lds_addr (matches our As[m][k] layout).
DEV void tdm_load_tile_2d(unsigned int lds_addr, const void* gaddr,
                          unsigned int tile_d0, unsigned int rows, unsigned int ld_elems) {
  unsigned long long ga = (unsigned long long)gaddr;
  u32x4 g0;
  g0[0] = 1u;                                              // count=1, user descriptor
  g0[1] = lds_addr;                                        // lds_addr (bytes)
  g0[2] = (unsigned int)(ga & 0xffffffffu);                // global_addr[31:0]
  g0[3] = (unsigned int)((ga >> 32) & 0x01ffffffu)         // global_addr[56:32]
          | (2u << 30);                                    // type=2 ("image")
  i32x8 g1;
  g1[0] = (int)(1u << 16);                                 // data_size=1 (2 bytes)
  g1[1] = (int)((tile_d0 & 0xffffu) << 16);                // tensor_dim0[15:0] = tile_d0
  g1[2] = (int)((tile_d0 >> 16) | ((rows & 0xffffu) << 16)); // dim0 hi | tensor_dim1 lo
  g1[3] = (int)((rows >> 16) | (tile_d0 << 16));           // dim1 hi | tile_dim0
  g1[4] = (int)rows;                                       // tile_dim1 (tile_dim2=0)
  g1[5] = (int)ld_elems;                                   // tensor_dim0_stride[31:0]
  g1[6] = 0;                                               // stride hi | dim1_stride lo
  g1[7] = 0;
  i32x4 z4 = {0, 0, 0, 0};
  i32x8 z8 = {0, 0, 0, 0, 0, 0, 0, 0};
  __builtin_amdgcn_tensor_load_to_lds(g0, g1, z4, z4, z8, 0);
}
template<int CNT>
DEV void wait_tensorcnt() { __builtin_amdgcn_s_wait_tensorcnt(CNT); }

// ---------------------------------------------------------------- elementwise
__global__ void cvt_f32_bf16(const float* __restrict__ s, unsigned short* __restrict__ d, int n) {
  for (int i = blockIdx.x * blockDim.x + threadIdx.x; i < n; i += gridDim.x * blockDim.x)
    d[i] = f32_to_bf16(s[i]);
}

__global__ void swiglu_kernel(const float* __restrict__ H, unsigned short* __restrict__ S,
                              int rows, int F) {
  int total = rows * F;
  for (int i = blockIdx.x * blockDim.x + threadIdx.x; i < total; i += gridDim.x * blockDim.x) {
    int r = i / F, c = i - r * F;
    float g = H[(size_t)r * 2 * F + c];
    float u = H[(size_t)r * 2 * F + F + c];
    S[i] = f32_to_bf16(silu_f(g) * u);
  }
}

__global__ void reduce_splitk(const float* __restrict__ part, float* __restrict__ outp,
                              int n, int z) {
  for (int i = blockIdx.x * blockDim.x + threadIdx.x; i < n; i += gridDim.x * blockDim.x) {
    float s = 0.f;
    for (int zz = 0; zz < z; ++zz) s += part[(size_t)zz * n + i];
    outp[i] = s;
  }
}

// ---------------------------------------------------------------- router
__global__ __launch_bounds__(64)
void router_kernel(const float* __restrict__ x, const float* __restrict__ gate_w,
                   const float* __restrict__ bias, float* __restrict__ combine,
                   float* __restrict__ lse2) {
  const int n = blockIdx.x;
  const int e = threadIdx.x;             // 0..63
  __shared__ float s_logit[64], s_aff[64], s_score[64];
  const float* xr = x + (size_t)n * 1024;
  const float* gr = gate_w + (size_t)e * 1024;
  float acc = 0.f;
  for (int d = 0; d < 1024; d += 4) {
    float4 xv = *(const float4*)&xr[d];
    float4 gv = *(const float4*)&gr[d];
    acc += xv.x * gv.x + xv.y * gv.y + xv.z * gv.z + xv.w * gv.w;
  }
  s_logit[e] = acc;
  float aff = 1.f / (1.f + __expf(-acc));
  s_aff[e] = aff;
  s_score[e] = aff + bias[e];
  combine[(size_t)n * 64 + e] = 0.f;
  __syncthreads();
  if (e == 0) {
    float m = s_logit[0];
    for (int i = 1; i < 64; ++i) m = fmaxf(m, s_logit[i]);
    float ssum = 0.f;
    for (int i = 0; i < 64; ++i) ssum += __expf(s_logit[i] - m);
    float lse = m + __logf(ssum);
    lse2[n] = lse * lse;
    int   sel[8];
    float asum = 0.f;
    for (int k = 0; k < 8; ++k) {
      int arg = 0; float best = -3.4e38f;
      for (int i = 0; i < 64; ++i)
        if (s_score[i] > best) { best = s_score[i]; arg = i; }
      sel[k] = arg; asum += s_aff[arg]; s_score[arg] = -3.4e38f;
    }
    float inv = 1.f / (asum + 1e-9f);
    for (int k = 0; k < 8; ++k)
      combine[(size_t)n * 64 + sel[k]] = s_aff[sel[k]] * inv;
  }
}

__global__ void zloss_kernel(const float* __restrict__ lse2, float* __restrict__ outz, int n) {
  __shared__ float red[256];
  int tid = threadIdx.x;
  float s = 0.f;
  for (int i = tid; i < n; i += 256) s += lse2[i];
  red[tid] = s; __syncthreads();
  for (int off = 128; off > 0; off >>= 1) {
    if (tid < off) red[tid] += red[tid + off];
    __syncthreads();
  }
  if (tid == 0) outz[0] = 0.001f * red[0] / (float)n;
}

// ---------------------------------------------------------------- WMMA GEMM
// C[M,N] = A[M,K](bf16,row) x B (bf16).  A tiles staged by TDM (wave 0, TENSORcnt).
// BT=1: B given as Bt[N,K] row-major, staged with async-to-LDS + double buffering.
// BT=0: B[K,N], sync transpose into LDS.
// Block: 128 threads = 4 waves; tile 64x128, BK=32; wave computes 32x64 (8 WMMA/step).
// gridDim.z = split-K factor; partial z written at C + z*M*ldc.
template<bool BT, bool ACCUM>
__global__ __launch_bounds__(128)
void gemm_bf16(const unsigned short* __restrict__ A, const unsigned short* __restrict__ B,
               float* __restrict__ C, int M, int N, int K, int lda, int ldb, int ldc) {
  __shared__ __align__(16) unsigned short As[2][64 * 32];    // [m][k]
  __shared__ __align__(16) unsigned short Bs[2][128 * 32];   // [n][k]
  const int tid = threadIdx.x, lane = tid & 31, wave = tid >> 5;
  const int wm = wave & 1, wn = wave >> 1;
  const int m0 = blockIdx.x * 64, n0 = blockIdx.y * 128;
  const int r15 = lane & 15, kh = lane >> 4;

  const int Kc = K / gridDim.z;
  const int kbeg = blockIdx.z * Kc;
  C += (size_t)blockIdx.z * (size_t)M * (size_t)ldc;

  const v8f zero = {0.f, 0.f, 0.f, 0.f, 0.f, 0.f, 0.f, 0.f};
  v8f acc[2][4] = {{zero, zero, zero, zero}, {zero, zero, zero, zero}};

  auto loadA_tdm = [&](int buf, int k0) {                    // 1 TDM op, wave 0 only
    if (wave == 0)
      tdm_load_tile_2d(lds_off_u32(&As[buf][0]),
                       &A[(size_t)m0 * lda + k0], 32u, 64u, (unsigned int)lda);
  };
  auto loadB_async = [&](int buf, int k0) {                  // 4 asyncs / thread
#pragma unroll
    for (int it = 0; it < 4; ++it) {
      const int c = tid + it * 128;
      const int nr = c >> 2, kc = (c & 3) << 3;
      async_copy_g2l_b128(lds_off_u32(&Bs[buf][nr * 32 + kc]),
                          &B[(size_t)(n0 + nr) * ldb + (k0 + kc)]);
    }
  };

  const int nk = Kc >> 5;
  if (BT) { loadA_tdm(0, kbeg); loadB_async(0, kbeg); }

  for (int kt = 0; kt < nk; ++kt) {
    const int k0 = kbeg + kt * 32;
    int buf;
    if (BT) {
      buf = kt & 1;
      if (kt + 1 < nk) {
        loadA_tdm((kt + 1) & 1, k0 + 32);
        loadB_async((kt + 1) & 1, k0 + 32);
        wait_tensorcnt<1>();                 // tile kt's TDM retired; kt+1 in flight
        WAIT_ASYNCCNT("0x4");                // tile kt's B resident
      } else {
        wait_tensorcnt<0>();
        WAIT_ASYNCCNT("0x0");
      }
      __syncthreads();
    } else {
      buf = 0;
      __syncthreads();                       // protect previous iteration's reads
      loadA_tdm(0, k0);
#pragma unroll
      for (int it = 0; it < 4; ++it) {       // B[k][n] -> Bs[n][k] transpose
        const int c = tid + it * 128;
        const int kr = c >> 4, nc = (c & 15) << 3;
        Pack8 t; t.q = *(const uint4*)&B[(size_t)(k0 + kr) * ldb + (n0 + nc)];
#pragma unroll
        for (int j = 0; j < 8; ++j) Bs[0][(nc + j) * 32 + kr] = t.h[j];
      }
      if (k0 + 32 < kbeg + Kc)
        __builtin_prefetch(&B[(size_t)(k0 + 32 + (tid >> 4)) * ldb + n0], 0, 1);
      wait_tensorcnt<0>();
      __syncthreads();
    }

    const unsigned short* as = As[buf];
    const unsigned short* bs = Bs[buf];
    FragB16 fa[2], fb[4];
#pragma unroll
    for (int tm = 0; tm < 2; ++tm) {         // A: lane row = r15, K = kh*16 + j
      const int row = wm * 32 + tm * 16 + r15;
      fa[tm].q[0] = *(const uint4*)&as[row * 32 + kh * 16];
      fa[tm].q[1] = *(const uint4*)&as[row * 32 + kh * 16 + 8];
    }
#pragma unroll
    for (int tn = 0; tn < 4; ++tn) {         // B: lane col = r15, K = kh*16 + j
      const int col = wn * 64 + tn * 16 + r15;
      fb[tn].q[0] = *(const uint4*)&bs[col * 32 + kh * 16];
      fb[tn].q[1] = *(const uint4*)&bs[col * 32 + kh * 16 + 8];
    }
#pragma unroll
    for (int tm = 0; tm < 2; ++tm)
#pragma unroll
      for (int tn = 0; tn < 4; ++tn)
        acc[tm][tn] = __builtin_amdgcn_wmma_f32_16x16x32_bf16(
            false, fa[tm].v, false, fb[tn].v, (short)0, acc[tm][tn], false, false);

    if (BT) __syncthreads();                 // reads done before buffer is re-filled
  }

#pragma unroll
  for (int tm = 0; tm < 2; ++tm)
#pragma unroll
    for (int tn = 0; tn < 4; ++tn) {
      const int nn = n0 + wn * 64 + tn * 16 + r15;
      const int mb = m0 + wm * 32 + tm * 16 + kh * 8;    // D: VGPR i -> row kh*8+i
#pragma unroll
      for (int i = 0; i < 8; ++i) {
        if (ACCUM) C[(size_t)(mb + i) * ldc + nn] += acc[tm][tn][i];
        else       C[(size_t)(mb + i) * ldc + nn]  = acc[tm][tn][i];
      }
    }
}

// ------------------------------------------------- batched expert GU + fused SwiGLU*combine
// lat[1024,256] x exp_gu[e][256,512]; block computes matching g and u 64x64 tiles,
// epilogue: hid[n, e*256+f] = silu(g)*u*combine[n,e]  (bf16)
__global__ __launch_bounds__(128)
void gemm_moe_gu(const unsigned short* __restrict__ lat, const unsigned short* __restrict__ exp_gu,
                 const float* __restrict__ combine, unsigned short* __restrict__ hid) {
  __shared__ __align__(16) unsigned short As[64 * 32];
  __shared__ __align__(16) unsigned short Bg[64 * 32];
  __shared__ __align__(16) unsigned short Bu[64 * 32];
  const int tid = threadIdx.x, lane = tid & 31, wave = tid >> 5;
  const int wm = wave & 1, wn = wave >> 1;
  const int m0 = blockIdx.x * 64;       // token tile
  const int c0 = blockIdx.y * 64;       // F-column tile (F=256)
  const int e  = blockIdx.z;
  const int r15 = lane & 15, kh = lane >> 4;
  const unsigned short* Be = exp_gu + (size_t)e * 256 * 512;

  const v8f zero = {0.f, 0.f, 0.f, 0.f, 0.f, 0.f, 0.f, 0.f};
  v8f accg[2][2] = {{zero, zero}, {zero, zero}};
  v8f accu[2][2] = {{zero, zero}, {zero, zero}};

  for (int k0 = 0; k0 < 256; k0 += 32) {
    __syncthreads();
    if (wave == 0)                                     // A tile via TDM
      tdm_load_tile_2d(lds_off_u32(&As[0]),
                       &lat[(size_t)m0 * 256 + k0], 32u, 64u, 256u);
#pragma unroll
    for (int it = 0; it < 2; ++it) {                   // Bg/Bu transpose into LDS
      const int c = tid + it * 128;
      const int kr = c >> 3, nc = (c & 7) << 3;
      Pack8 tg, tu;
      tg.q = *(const uint4*)&Be[(size_t)(k0 + kr) * 512 + (c0 + nc)];
      tu.q = *(const uint4*)&Be[(size_t)(k0 + kr) * 512 + (256 + c0 + nc)];
#pragma unroll
      for (int j = 0; j < 8; ++j) {
        Bg[(nc + j) * 32 + kr] = tg.h[j];
        Bu[(nc + j) * 32 + kr] = tu.h[j];
      }
    }
    wait_tensorcnt<0>();
    __syncthreads();

    FragB16 fa[2], fg[2], fu[2];
#pragma unroll
    for (int tm = 0; tm < 2; ++tm) {
      const int row = wm * 32 + tm * 16 + r15;
      fa[tm].q[0] = *(const uint4*)&As[row * 32 + kh * 16];
      fa[tm].q[1] = *(const uint4*)&As[row * 32 + kh * 16 + 8];
    }
#pragma unroll
    for (int tn = 0; tn < 2; ++tn) {
      const int col = wn * 32 + tn * 16 + r15;
      fg[tn].q[0] = *(const uint4*)&Bg[col * 32 + kh * 16];
      fg[tn].q[1] = *(const uint4*)&Bg[col * 32 + kh * 16 + 8];
      fu[tn].q[0] = *(const uint4*)&Bu[col * 32 + kh * 16];
      fu[tn].q[1] = *(const uint4*)&Bu[col * 32 + kh * 16 + 8];
    }
#pragma unroll
    for (int tm = 0; tm < 2; ++tm)
#pragma unroll
      for (int tn = 0; tn < 2; ++tn) {
        accg[tm][tn] = __builtin_amdgcn_wmma_f32_16x16x32_bf16(
            false, fa[tm].v, false, fg[tn].v, (short)0, accg[tm][tn], false, false);
        accu[tm][tn] = __builtin_amdgcn_wmma_f32_16x16x32_bf16(
            false, fa[tm].v, false, fu[tn].v, (short)0, accu[tm][tn], false, false);
      }
  }

#pragma unroll
  for (int tm = 0; tm < 2; ++tm)
#pragma unroll
    for (int tn = 0; tn < 2; ++tn) {
      const int f  = c0 + wn * 32 + tn * 16 + r15;
      const int mb = m0 + wm * 32 + tm * 16 + kh * 8;
#pragma unroll
      for (int i = 0; i < 8; ++i) {
        const int m = mb + i;
        const float cw = combine[(size_t)m * 64 + e];
        const float g = accg[tm][tn][i], u = accu[tm][tn][i];
        hid[(size_t)m * 16384 + e * 256 + f] = f32_to_bf16(silu_f(g) * u * cw);
      }
    }
}

// ---------------------------------------------------------------- launcher
extern "C" void kernel_launch(void* const* d_in, const int* in_sizes, int n_in,
                              void* d_out, int out_size, void* d_ws, size_t ws_size,
                              hipStream_t stream) {
  constexpr int N = 1024, D = 1024, E = 64, L = 256, F = 256, FS = 1024;
  constexpr int SPLITK = 8;
  const float* x             = (const float*)d_in[0];
  const float* gate_w        = (const float*)d_in[1];
  const float* expert_bias   = (const float*)d_in[2];
  const float* down_w        = (const float*)d_in[3];
  const float* up_w          = (const float*)d_in[4];
  const float* shared_gu_w   = (const float*)d_in[5];
  const float* shared_down_w = (const float*)d_in[6];
  const float* exp_gu_w      = (const float*)d_in[7];
  const float* exp_down_w    = (const float*)d_in[8];
  float* out = (float*)d_out;

  char* wp = (char*)d_ws;
  auto alloc = [&](size_t bytes) -> void* {
    void* p = (void*)wp; wp += (bytes + 255) & ~(size_t)255; return p;
  };
  unsigned short* xbf   = (unsigned short*)alloc((size_t)N * D * 2);
  unsigned short* sgubf = (unsigned short*)alloc((size_t)2 * FS * D * 2);
  unsigned short* sdbf  = (unsigned short*)alloc((size_t)D * FS * 2);
  unsigned short* dwbf  = (unsigned short*)alloc((size_t)L * D * 2);
  unsigned short* upbf  = (unsigned short*)alloc((size_t)D * L * 2);
  unsigned short* egubf = (unsigned short*)alloc((size_t)E * L * 2 * F * 2);
  unsigned short* edbf  = (unsigned short*)alloc((size_t)E * F * L * 2);
  float* combine        = (float*)alloc((size_t)N * E * 4);
  float* lse2           = (float*)alloc((size_t)N * 4);
  float* Hs             = (float*)alloc((size_t)N * 2 * FS * 4);
  unsigned short* Sbf   = (unsigned short*)alloc((size_t)N * FS * 2);
  float* latf           = (float*)alloc((size_t)N * L * 4);
  unsigned short* latbf = (unsigned short*)alloc((size_t)N * L * 2);
  unsigned short* hid   = (unsigned short*)alloc((size_t)N * E * F * 2);
  float* rlpart         = (float*)alloc((size_t)SPLITK * N * L * 4);
  float* rlf            = (float*)alloc((size_t)N * L * 4);
  unsigned short* rlbf  = (unsigned short*)alloc((size_t)N * L * 2);

  auto cvt = [&](const float* s, unsigned short* dst, int n) {
    int g = (n + 255) / 256; if (g > 4096) g = 4096;
    cvt_f32_bf16<<<g, 256, 0, stream>>>(s, dst, n);
  };
  cvt(x, xbf, N * D);
  cvt(shared_gu_w, sgubf, 2 * FS * D);
  cvt(shared_down_w, sdbf, D * FS);
  cvt(down_w, dwbf, L * D);
  cvt(up_w, upbf, D * L);
  cvt(exp_gu_w, egubf, E * L * 2 * F);
  cvt(exp_down_w, edbf, E * F * L);

  // Router + z-loss (z_loss stored as the last output element)
  router_kernel<<<N, 64, 0, stream>>>(x, gate_w, expert_bias, combine, lse2);
  zloss_kernel<<<1, 256, 0, stream>>>(lse2, out + (size_t)N * D, N);

  // Shared expert: Hs = x @ shared_gu^T  -> swiglu -> out = S @ shared_down^T
  gemm_bf16<true, false><<<dim3(N / 64, (2 * FS) / 128, 1), 128, 0, stream>>>(
      xbf, sgubf, Hs, N, 2 * FS, D, D, D, 2 * FS);
  swiglu_kernel<<<(N * FS + 255) / 256, 256, 0, stream>>>(Hs, Sbf, N, FS);
  gemm_bf16<true, false><<<dim3(N / 64, D / 128, 1), 128, 0, stream>>>(
      Sbf, sdbf, out, N, D, FS, FS, FS, D);

  // Latent path: lat = x @ down^T
  gemm_bf16<true, false><<<dim3(N / 64, L / 128, 1), 128, 0, stream>>>(
      xbf, dwbf, latf, N, L, D, D, D, L);
  cvt(latf, latbf, N * L);

  // Batched expert GU with fused SwiGLU * combine-weight
  gemm_moe_gu<<<dim3(N / 64, F / 64, E), 128, 0, stream>>>(latbf, egubf, combine, hid);

  // routed_lat = hid[N, E*F] @ exp_down_flat[E*F, L]  (split-K, deterministic reduce)
  gemm_bf16<false, false><<<dim3(N / 64, L / 128, SPLITK), 128, 0, stream>>>(
      hid, edbf, rlpart, N, L, E * F, E * F, L, L);
  reduce_splitk<<<(N * L + 255) / 256, 256, 0, stream>>>(rlpart, rlf, N * L, SPLITK);
  cvt(rlf, rlbf, N * L);

  // out += routed_lat @ up^T
  gemm_bf16<true, true><<<dim3(N / 64, D / 128, 1), 128, 0, stream>>>(
      rlbf, upbf, out, N, D, L, L, L, D);
}
